// Model_GAI_3281355014320
// MI455X (gfx1250) — matrix operand order repaired and checked by reading, not compiled, not run on hardware
//
#include <hip/hip_runtime.h>
#include <hip/hip_bf16.h>
#include <math.h>

// ---------- problem constants ----------
#define BB2     2
#define T_LEN   4096
#define D_DIM   2048
#define K_MU    8
#define C_OUT   21
#define KTOT    (3 * D_DIM)      // 6144
#define M_TOT   (BB2 * T_LEN)    // 8192

// ---------- WMMA types ----------
typedef __attribute__((ext_vector_type(16))) __bf16 v16bf;
typedef __attribute__((ext_vector_type(8)))  float  v8f;
typedef __attribute__((ext_vector_type(4)))  int    v4i_;

#ifndef __has_builtin
#define __has_builtin(x) 0
#endif

#if __has_builtin(__builtin_amdgcn_global_load_async_to_lds_b128)
#define USE_ASYNC 1
#else
#define USE_ASYNC 0
#endif

#if USE_ASYNC
__device__ __forceinline__ void async_cp16(const void* gsrc, void* ldst) {
    __builtin_amdgcn_global_load_async_to_lds_b128(
        (__attribute__((address_space(1))) v4i_*)gsrc,
        (__attribute__((address_space(3))) v4i_*)ldst,
        0, 0);
}
template <int N>
__device__ __forceinline__ void wait_async() {
#if __has_builtin(__builtin_amdgcn_s_wait_asynccnt)
    __builtin_amdgcn_s_wait_asynccnt(N);
#else
    asm volatile("s_wait_asynccnt %0" :: "n"(N) : "memory");
#endif
}
#endif

static __device__ __forceinline__ unsigned short f2bf(float f) {
    union { float f; unsigned int u; } a; a.f = f;
    unsigned int u = a.u;
    unsigned int r = (u + 0x7FFFu + ((u >> 16) & 1u)) >> 16;   // round-nearest-even
    return (unsigned short)r;
}

// ============================================================
// 1) pad + convert x -> bf16 [B][T+2][D]
// ============================================================
__global__ __launch_bounds__(256)
void pad_x_kernel(const float* __restrict__ x, unsigned short* __restrict__ xb) {
    size_t idx = (size_t)blockIdx.x * 256 + threadIdx.x;
    size_t total = (size_t)BB2 * (T_LEN + 2) * D_DIM;
    if (idx >= total) return;
    int   i  = (int)(idx % D_DIM);
    size_t r = idx / D_DIM;
    int   tp = (int)(r % (T_LEN + 2));
    int   b  = (int)(r / (T_LEN + 2));
    float v = 0.f;
    if (tp > 0 && tp < T_LEN + 1)
        v = x[((size_t)b * T_LEN + (tp - 1)) * D_DIM + i];
    xb[idx] = f2bf(v);
}

// ============================================================
// 2) repack conv1_w (OIH [c][i][k]) -> bf16 [KTOT/32][N=2048][32]
// ============================================================
__global__ __launch_bounds__(256)
void repack_w_kernel(const float* __restrict__ w, unsigned short* __restrict__ wb2) {
    size_t idx = (size_t)blockIdx.x * 256 + threadIdx.x;
    size_t total = (size_t)(KTOT / 32) * D_DIM * 32;
    if (idx >= total) return;
    int j = (int)(idx & 31);
    int c = (int)((idx >> 5) & (D_DIM - 1));
    int s = (int)(idx >> 16);            // 2048*32 = 65536 per K-slab
    int kk   = s * 32 + j;
    int kcv  = kk >> 11;                 // conv tap 0..2
    int i    = kk & (D_DIM - 1);
    wb2[idx] = f2bf(w[((size_t)c * D_DIM + i) * 3 + kcv]);
}

// ============================================================
// 3) implicit-GEMM conv: feat = relu(A @ B + bias), bf16 WMMA
//    128x128 tile / workgroup, 8 waves, wave = 32x64,
//    double-buffered LDS, async global->LDS when available
// ============================================================
#define TM 128
#define TN 128
#define KC 32
#define NSTAGE (KTOT / KC)

__global__ __launch_bounds__(256)
void conv_gemm_kernel(const unsigned short* __restrict__ xb,   // [B][T+2][D]
                      const unsigned short* __restrict__ wb2,  // [KTOT/32][2048][32]
                      const float* __restrict__ bias,
                      float* __restrict__ feat)                // [M][2048] fp32
{
    __shared__ unsigned short As[2][TM][KC];   // [row][k]   8KB each
    __shared__ unsigned short Bs[2][TN][KC];   // [col][k]   8KB each

    const int tid  = threadIdx.x;
    const int lane = tid & 31;
    const int w    = tid >> 5;
    const int n0   = blockIdx.x * TN;
    const int m0   = blockIdx.y * TM;
    const int bb   = m0 / T_LEN;
    const int t0   = m0 % T_LEN;

    const int wm  = (w & 3) * 32;
    const int wn  = (w >> 2) * 64;
    const int h   = lane >> 4;
    const int l15 = lane & 15;

    v8f acc[2][4] = {};

    // per-thread chunk geometry: chunks c = tid and tid+256 (512 x 16B per tile)
    const int cA0 = tid,        cA1 = tid + 256;
    const int rA0 = cA0 >> 2,   pA0 = cA0 & 3;
    const int rA1 = cA1 >> 2,   pA1 = cA1 & 3;

#if USE_ASYNC
    auto issue_stage = [&](int s, int buf) {
        const int kcv = (s * KC) >> 11;
        const int i0  = (s * KC) & (D_DIM - 1);
        const unsigned short* gA =
            xb + ((size_t)bb * (T_LEN + 2) + (size_t)(t0 + kcv)) * D_DIM + i0;
        const unsigned short* gB = wb2 + ((size_t)s * D_DIM + n0) * KC;
        unsigned short* sA = &As[buf][0][0];
        unsigned short* sB = &Bs[buf][0][0];
        async_cp16(gA + (size_t)rA0 * D_DIM + pA0 * 8, sA + (size_t)cA0 * 8);
        async_cp16(gB + (size_t)cA0 * 8,               sB + (size_t)cA0 * 8);
        async_cp16(gA + (size_t)rA1 * D_DIM + pA1 * 8, sA + (size_t)cA1 * 8);
        async_cp16(gB + (size_t)cA1 * 8,               sB + (size_t)cA1 * 8);
    };
#else
    uint4 rgA[2], rgB[2];
    auto load_regs = [&](int s) {
        const int kcv = (s * KC) >> 11;
        const int i0  = (s * KC) & (D_DIM - 1);
        const uint4* gA = reinterpret_cast<const uint4*>(
            xb + ((size_t)bb * (T_LEN + 2) + (size_t)(t0 + kcv)) * D_DIM + i0);
        const uint4* gB = reinterpret_cast<const uint4*>(
            wb2 + ((size_t)s * D_DIM + n0) * KC);
        rgA[0] = gA[(size_t)rA0 * (D_DIM / 8) + pA0];
        rgA[1] = gA[(size_t)rA1 * (D_DIM / 8) + pA1];
        rgB[0] = gB[cA0];
        rgB[1] = gB[cA1];
    };
    auto store_regs = [&](int buf) {
        uint4* sA = reinterpret_cast<uint4*>(&As[buf][0][0]);
        uint4* sB = reinterpret_cast<uint4*>(&Bs[buf][0][0]);
        sA[cA0] = rgA[0]; sA[cA1] = rgA[1];
        sB[cA0] = rgB[0]; sB[cA1] = rgB[1];
    };
#endif

    auto compute_stage = [&](int buf) {
        union Frag { uint4 q[2]; v16bf v; };
        Frag a[2], b[4];
        #pragma unroll
        for (int mt = 0; mt < 2; ++mt) {
            int row = wm + mt * 16 + l15;
            a[mt].q[0] = *reinterpret_cast<const uint4*>(&As[buf][row][h * 8]);
            a[mt].q[1] = *reinterpret_cast<const uint4*>(&As[buf][row][16 + h * 8]);
        }
        #pragma unroll
        for (int nt = 0; nt < 4; ++nt) {
            int col = wn + nt * 16 + l15;
            b[nt].q[0] = *reinterpret_cast<const uint4*>(&Bs[buf][col][h * 8]);
            b[nt].q[1] = *reinterpret_cast<const uint4*>(&Bs[buf][col][16 + h * 8]);
        }
        #pragma unroll
        for (int mt = 0; mt < 2; ++mt)
            #pragma unroll
            for (int nt = 0; nt < 4; ++nt)
                acc[mt][nt] = __builtin_amdgcn_wmma_f32_16x16x32_bf16(
                    false, a[mt].v, false, b[nt].v,
                    (short)0, acc[mt][nt], false, false);
    };

#if USE_ASYNC
    issue_stage(0, 0);
    for (int s = 0; s < NSTAGE; ++s) {
        const int buf = s & 1;
        if (s + 1 < NSTAGE) {
            issue_stage(s + 1, buf ^ 1);   // overlap next-slab DMA with compute
            wait_async<4>();               // retire current slab, keep next 4 in flight
        } else {
            wait_async<0>();
        }
        __syncthreads();                   // all waves' slab data visible
        compute_stage(buf);
        __syncthreads();                   // done reading buf before it is refilled
    }
#else
    load_regs(0);
    for (int s = 0; s < NSTAGE; ++s) {
        const int buf = s & 1;
        store_regs(buf);
        __syncthreads();
        if (s + 1 < NSTAGE) load_regs(s + 1);  // prefetch overlaps compute
        compute_stage(buf);
    }
#endif

    // epilogue: bias + relu; C layout: VGPR r -> row r + 8*h, lane&15 -> col
    #pragma unroll
    for (int mt = 0; mt < 2; ++mt)
        #pragma unroll
        for (int nt = 0; nt < 4; ++nt) {
            int colg = n0 + wn + nt * 16 + l15;
            float bv = bias[colg];
            #pragma unroll
            for (int r = 0; r < 8; ++r) {
                int rowg = m0 + wm + mt * 16 + r + 8 * h;
                float v = acc[mt][nt][r] + bv;
                feat[(size_t)rowg * D_DIM + colg] = v > 0.f ? v : 0.f;
            }
        }
}

// ============================================================
// 4) row inverse-norms: rinv[row] = 1/(||src[row,:]||2 + 1e-9)
// ============================================================
__global__ __launch_bounds__(256)
void rownorm_kernel(const float* __restrict__ src, float* __restrict__ rinv, int ncols) {
    int row = blockIdx.x;
    __shared__ float red[256];
    float s = 0.f;
    for (int d = threadIdx.x; d < ncols; d += 256) {
        float v = src[(size_t)row * ncols + d];
        s += v * v;
    }
    red[threadIdx.x] = s; __syncthreads();
    for (int st = 128; st > 0; st >>= 1) {
        if (threadIdx.x < st) red[threadIdx.x] += red[threadIdx.x + st];
        __syncthreads();
    }
    if (threadIdx.x == 0) rinv[row] = 1.f / (sqrtf(red[0]) + 1e-9f);
}

// normalize mu rows into nmu (16 rows x 2048)
__global__ __launch_bounds__(256)
void normrows_kernel(const float* __restrict__ mu, float* __restrict__ nmu) {
    int row = blockIdx.x;
    __shared__ float red[256];
    float s = 0.f;
    for (int d = threadIdx.x; d < D_DIM; d += 256) {
        float v = mu[(size_t)row * D_DIM + d];
        s += v * v;
    }
    red[threadIdx.x] = s; __syncthreads();
    for (int st = 128; st > 0; st >>= 1) {
        if (threadIdx.x < st) red[threadIdx.x] += red[threadIdx.x + st];
        __syncthreads();
    }
    __shared__ float invs;
    if (threadIdx.x == 0) invs = 1.f / (sqrtf(red[0]) + 1e-9f);
    __syncthreads();
    float inv = invs;
    for (int d = threadIdx.x; d < D_DIM; d += 256)
        nmu[(size_t)row * D_DIM + d] = mu[(size_t)row * D_DIM + d] * inv;
}

__global__ __launch_bounds__(256)
void mu_init_kernel(const float* __restrict__ mu_param, float* __restrict__ mu0) {
    int idx = blockIdx.x * 256 + threadIdx.x;
    if (idx < BB2 * K_MU * D_DIM) mu0[idx] = mu_param[idx % (K_MU * D_DIM)];
}

__global__ void zero_kernel(float* p, int n) {
    int idx = blockIdx.x * 256 + threadIdx.x;
    if (idx < n) p[idx] = 0.f;
}

// ============================================================
// 5) z = softmax_k(5 * rinv[t] * <nmu[k], X[t]>); optional col-sums
//    one wave per t (8 waves / block)
// ============================================================
__global__ __launch_bounds__(256)
void zsoft_kernel(const float* __restrict__ X, const float* __restrict__ rinv,
                  const float* __restrict__ nmu, float* __restrict__ z,
                  float* __restrict__ zsum, int addsums)
{
    int w = threadIdx.x >> 5, lane = threadIdx.x & 31;
    int tg = blockIdx.x * 8 + w;
    int b = tg / T_LEN, t = tg % T_LEN;
    const float* xr = X + ((size_t)b * T_LEN + t) * D_DIM;
    const float* mb = nmu + (size_t)b * K_MU * D_DIM;

    float s[K_MU] = {0, 0, 0, 0, 0, 0, 0, 0};
    for (int d = lane; d < D_DIM; d += 32) {
        float xv = xr[d];
        #pragma unroll
        for (int k = 0; k < K_MU; ++k) s[k] += xv * mb[(size_t)k * D_DIM + d];
    }
    #pragma unroll
    for (int k = 0; k < K_MU; ++k)
        for (int off = 16; off > 0; off >>= 1) s[k] += __shfl_xor(s[k], off, 32);

    if (lane == 0) {
        float scale = 5.f * rinv[b * T_LEN + t];
        float mx = -1e30f;
        #pragma unroll
        for (int k = 0; k < K_MU; ++k) { s[k] *= scale; mx = fmaxf(mx, s[k]); }
        float sum = 0.f, e[K_MU];
        #pragma unroll
        for (int k = 0; k < K_MU; ++k) { e[k] = expf(s[k] - mx); sum += e[k]; }
        #pragma unroll
        for (int k = 0; k < K_MU; ++k) {
            float zz = e[k] / sum;
            z[((size_t)b * K_MU + k) * T_LEN + t] = zz;
            if (addsums) atomicAdd(&zsum[b * K_MU + k], zz);
        }
    }
}

// ============================================================
// 6) mu_out[b,k,d] = (1/(zsum+1e-9)) * sum_t z[b,k,t]*X[b,t,d]
// ============================================================
__global__ __launch_bounds__(256)
void mu_update_kernel(const float* __restrict__ X, const float* __restrict__ z,
                      const float* __restrict__ zsum, float* __restrict__ mu_out)
{
    int dchunk = blockIdx.x & 7;
    int k = (blockIdx.x >> 3) & 7;
    int b = blockIdx.x >> 6;
    int d = dchunk * 256 + threadIdx.x;
    const float* zr = z + ((size_t)b * K_MU + k) * T_LEN;
    float inv = 1.f / (zsum[b * K_MU + k] + 1e-9f);
    const float* xc = X + (size_t)b * T_LEN * D_DIM + d;
    float acc = 0.f;
    for (int t = 0; t < T_LEN; ++t) acc += zr[t] * xc[(size_t)t * D_DIM];
    mu_out[((size_t)b * K_MU + k) * D_DIM + d] = acc * inv;
}

// ============================================================
// 7) muC[b,k,c] = <mu[b,k], cls_w[c]>
// ============================================================
__global__ __launch_bounds__(256)
void muc_kernel(const float* __restrict__ mu, const float* __restrict__ clsw,
                float* __restrict__ muC)
{
    int c  = blockIdx.x % C_OUT;
    int kk = blockIdx.x / C_OUT;       // b*8+k
    __shared__ float red[256];
    float s = 0.f;
    for (int d = threadIdx.x; d < D_DIM; d += 256)
        s += mu[(size_t)kk * D_DIM + d] * clsw[(size_t)c * D_DIM + d];
    red[threadIdx.x] = s; __syncthreads();
    for (int st = 128; st > 0; st >>= 1) {
        if (threadIdx.x < st) red[threadIdx.x] += red[threadIdx.x + st];
        __syncthreads();
    }
    if (threadIdx.x == 0) muC[kk * C_OUT + c] = red[0];
}

// ============================================================
// 8) head: cas = sigmoid(feat @ cls_w^T)
//    r_cas = sigmoid(0.25 * sum_k z2*muC + 0.5 * (feat @ cls_w^T))
// ============================================================
__global__ __launch_bounds__(256)
void cls_kernel(const float* __restrict__ feat, const float* __restrict__ z2,
                const float* __restrict__ muC, const float* __restrict__ clsw,
                float* __restrict__ out)
{
    int m = blockIdx.x;
    int b = m / T_LEN, t = m % T_LEN;
    __shared__ float frow[D_DIM];
    __shared__ float zk[K_MU];
    for (int d = threadIdx.x; d < D_DIM; d += 256)
        frow[d] = feat[(size_t)m * D_DIM + d];
    if (threadIdx.x < K_MU)
        zk[threadIdx.x] = z2[((size_t)b * K_MU + threadIdx.x) * T_LEN + t];
    __syncthreads();

    int w = threadIdx.x >> 5, lane = threadIdx.x & 31;
    for (int c = w; c < C_OUT; c += 8) {
        const float* wr = clsw + (size_t)c * D_DIM;
        float s = 0.f;
        for (int d = lane; d < D_DIM; d += 32) s += frow[d] * wr[d];
        for (int off = 16; off > 0; off >>= 1) s += __shfl_xor(s, off, 32);
        if (lane == 0) {
            float rl = 0.f;
            #pragma unroll
            for (int k = 0; k < K_MU; ++k) rl += zk[k] * muC[(b * K_MU + k) * C_OUT + c];
            rl = 0.25f * rl + 0.5f * s;
            size_t oi = (size_t)m * C_OUT + c;
            out[oi] = 1.f / (1.f + expf(-s));
            out[(size_t)M_TOT * C_OUT + oi] = 1.f / (1.f + expf(-rl));
        }
    }
}

// ============================================================
// host launcher
// ============================================================
static inline size_t alignup(size_t x) { return (x + 255) & ~(size_t)255; }

extern "C" void kernel_launch(void* const* d_in, const int* in_sizes, int n_in,
                              void* d_out, int out_size, void* d_ws, size_t ws_size,
                              hipStream_t stream) {
    const float* x        = (const float*)d_in[0];
    const float* conv1_w  = (const float*)d_in[1];
    const float* conv1_b  = (const float*)d_in[2];
    const float* mu_param = (const float*)d_in[3];
    const float* cls_w    = (const float*)d_in[4];
    float* out = (float*)d_out;

    char* ws = (char*)d_ws;
    size_t off = 0;
    unsigned short* xb   = (unsigned short*)(ws + off); off += alignup((size_t)BB2 * (T_LEN + 2) * D_DIM * 2);
    unsigned short* wb2  = (unsigned short*)(ws + off); off += alignup((size_t)KTOT * D_DIM * 2);
    float* feat   = (float*)(ws + off); off += alignup((size_t)M_TOT * D_DIM * 4);
    float* rnx    = (float*)(ws + off); off += alignup((size_t)M_TOT * 4);
    float* rnf    = (float*)(ws + off); off += alignup((size_t)M_TOT * 4);
    float* muA    = (float*)(ws + off); off += alignup((size_t)BB2 * K_MU * D_DIM * 4);
    float* muB    = (float*)(ws + off); off += alignup((size_t)BB2 * K_MU * D_DIM * 4);
    float* nmu    = (float*)(ws + off); off += alignup((size_t)BB2 * K_MU * D_DIM * 4);
    float* zbuf   = (float*)(ws + off); off += alignup((size_t)BB2 * K_MU * T_LEN * 4);
    float* z2buf  = (float*)(ws + off); off += alignup((size_t)BB2 * K_MU * T_LEN * 4);
    float* zsum   = (float*)(ws + off); off += alignup((size_t)BB2 * K_MU * 4);
    float* muC    = (float*)(ws + off); off += alignup((size_t)BB2 * K_MU * C_OUT * 4);

    // 1-2) stage bf16 operands
    {
        size_t n = (size_t)BB2 * (T_LEN + 2) * D_DIM;
        pad_x_kernel<<<dim3((unsigned)((n + 255) / 256)), dim3(256), 0, stream>>>(x, xb);
        size_t n2 = (size_t)KTOT * D_DIM;
        repack_w_kernel<<<dim3((unsigned)((n2 + 255) / 256)), dim3(256), 0, stream>>>(conv1_w, wb2);
    }
    // 3) conv as WMMA GEMM
    conv_gemm_kernel<<<dim3(D_DIM / TN, M_TOT / TM), dim3(256), 0, stream>>>(xb, wb2, conv1_b, feat);

    // 4) row norms of x ; init mu
    rownorm_kernel<<<dim3(M_TOT), dim3(256), 0, stream>>>(x, rnx, D_DIM);
    mu_init_kernel<<<dim3((BB2 * K_MU * D_DIM + 255) / 256), dim3(256), 0, stream>>>(mu_param, muA);

    // 5-6) two EM iterations (muA -> muB -> muA)
    float* mus[3] = {muA, muB, muA};
    for (int it = 0; it < 2; ++it) {
        normrows_kernel<<<dim3(BB2 * K_MU), dim3(256), 0, stream>>>(mus[it], nmu);
        zero_kernel<<<dim3(1), dim3(256), 0, stream>>>(zsum, BB2 * K_MU);
        zsoft_kernel<<<dim3(M_TOT / 8), dim3(256), 0, stream>>>(x, rnx, nmu, zbuf, zsum, 1);
        mu_update_kernel<<<dim3(BB2 * K_MU * 8), dim3(256), 0, stream>>>(x, zbuf, zsum, mus[it + 1]);
    }

    // 7) random-walk softmax over features (z2), fused head prep
    rownorm_kernel<<<dim3(M_TOT), dim3(256), 0, stream>>>(feat, rnf, D_DIM);
    normrows_kernel<<<dim3(BB2 * K_MU), dim3(256), 0, stream>>>(muA, nmu);
    zsoft_kernel<<<dim3(M_TOT / 8), dim3(256), 0, stream>>>(feat, rnf, nmu, z2buf, zsum, 0);
    muc_kernel<<<dim3(BB2 * K_MU * C_OUT), dim3(256), 0, stream>>>(muA, cls_w, muC);

    // 8) classifier head + sigmoids
    cls_kernel<<<dim3(M_TOT), dim3(256), 0, stream>>>(feat, z2buf, muC, cls_w, out);

    (void)in_sizes; (void)n_in; (void)out_size; (void)ws_size;
}
